// DiffusionModule_8624294330888
// MI455X (gfx1250) — matrix operand order repaired and checked by reading, not compiled, not run on hardware
//
#include <hip/hip_runtime.h>
#include <hip/hip_bf16.h>

// ---------------- types ----------------
typedef __bf16 bf16_t;
typedef __attribute__((ext_vector_type(16))) __bf16 v16bf;
typedef __attribute__((ext_vector_type(8)))  float  v8f;

union AFrag { v16bf v; uint4 u[2]; };

#define TOK   1024
#define DS_   768
#define DP_   128
#define H_    16
#define DH_   48
#define DFF_  1536

// ---- gfx1250 async global->LDS copy via inline asm (ISA §15.18, tracked by ASYNCcnt) ----
// VDST = per-lane LDS byte address (low 32 bits of generic LDS pointer),
// VADDR = 64-bit global address, no SADDR ("off").
#define USE_ASYNC_ASM 1

__device__ __forceinline__ void copy16B_g2l(const bf16_t* g, bf16_t* l) {
#if USE_ASYNC_ASM
    unsigned lds_off = (unsigned)(size_t)(void*)l;                 // flat->LDS: addr[31:0]
    unsigned long long ga = (unsigned long long)(size_t)(const void*)g;
    asm volatile("global_load_async_to_lds_b128 %0, %1, off"
                 :: "v"(lds_off), "v"(ga) : "memory");
#else
    *reinterpret_cast<uint4*>(l) = *reinterpret_cast<const uint4*>(g);
#endif
}
__device__ __forceinline__ void async_wait_all() {
#if USE_ASYNC_ASM
    asm volatile("s_wait_asynccnt 0x0" ::: "memory");
#endif
}

// ---------------- f32 -> bf16 convert ----------------
__global__ void cvt_f32_bf16_k(const float* __restrict__ in, bf16_t* __restrict__ out, int n) {
    int i = blockIdx.x * 256 + threadIdx.x;
    if (i < n) out[i] = (bf16_t)in[i];
}

// ---------------- row LayerNorm (C = 768), optional bf16 / f32 outs ----------------
__global__ __launch_bounds__(256)
void ln_rows_k(const float* __restrict__ in, bf16_t* __restrict__ outb,
               float* __restrict__ outf, int C) {
    const int row = blockIdx.x;
    const float* x = in + (size_t)row * C;
    const int tid = threadIdx.x;
    float s = 0.f, q = 0.f;
    for (int c = tid; c < C; c += 256) { float v = x[c]; s += v; q += v * v; }
    __shared__ float rs[8], rq[8];
    for (int m = 1; m < 32; m <<= 1) { s += __shfl_xor(s, m, 32); q += __shfl_xor(q, m, 32); }
    int wave = tid >> 5, lane = tid & 31;
    if (lane == 0) { rs[wave] = s; rq[wave] = q; }
    __syncthreads();
    if (tid == 0) {
        float ts = 0.f, tq = 0.f;
        for (int i = 0; i < 8; ++i) { ts += rs[i]; tq += rq[i]; }
        rs[0] = ts; rq[0] = tq;
    }
    __syncthreads();
    float mean = rs[0] / C;
    float var  = rq[0] / C - mean * mean;
    float inv  = rsqrtf(var + 1e-5f);
    for (int c = tid; c < C; c += 256) {
        float v = (x[c] - mean) * inv;
        if (outb) outb[(size_t)row * C + c] = (bf16_t)v;
        if (outf) outf[(size_t)row * C + c] = v;
    }
}

// ---------------- WMMA bf16 GEMM: D = act(A@B + bias) ----------------
// A [M x K] bf16 row-major, B [K x N] bf16 row-major; tile 128x64x32; 8 waves.
template<int ACT, int OBF>   // ACT: 0=none 1=sigmoid ; OBF: 0=f32 out 1=bf16 out
__global__ __launch_bounds__(256)
void gemm_bf16_k(const bf16_t* __restrict__ A, const bf16_t* __restrict__ Bm,
                 const float* __restrict__ bias, void* __restrict__ Out,
                 int M, int Nn, int K) {
    __shared__ alignas(16) bf16_t lsA[128][32];
    __shared__ alignas(16) bf16_t lsB[64][32];   // transposed: [n][k]
    const int tid  = threadIdx.x;
    const int wave = tid >> 5, lane = tid & 31;
    const int m0 = blockIdx.y * 128, n0 = blockIdx.x * 64;

    v8f acc[4];
    for (int t = 0; t < 4; ++t) for (int i = 0; i < 8; ++i) acc[t][i] = 0.f;

    for (int k0 = 0; k0 < K; k0 += 32) {
        {   // A tile 128x32: each thread copies 16 contiguous bf16 (two async b128)
            int row = tid >> 1, half = tid & 1;
            const bf16_t* src = A + (size_t)(m0 + row) * K + k0 + 16 * half;
            bf16_t* dst = &lsA[row][16 * half];
            copy16B_g2l(src, dst);
            copy16B_g2l(src + 8, dst + 8);
        }
        {   // B tile 32x64 -> transposed LDS
            int kk = tid >> 3, seg = tid & 7;
            const bf16_t* src = Bm + (size_t)(k0 + kk) * Nn + n0 + 8 * seg;
            #pragma unroll
            for (int e = 0; e < 8; ++e) lsB[8 * seg + e][kk] = src[e];
        }
        async_wait_all();
        __syncthreads();
        AFrag af;
        {   // A fragment: lanes 0-15 hold K 0..7 / 16..23; lanes 16-31 hold K 8..15 / 24..31
            int r = 16 * wave + (lane & 15), kh = lane >> 4;
            af.u[0] = *reinterpret_cast<const uint4*>(&lsA[r][8 * kh]);
            af.u[1] = *reinterpret_cast<const uint4*>(&lsA[r][8 * kh + 16]);
        }
        #pragma unroll
        for (int t = 0; t < 4; ++t) {
            AFrag bfg;
            int c = 16 * t + (lane & 15), kb2 = 8 * (lane >> 4);
            bfg.u[0] = *reinterpret_cast<const uint4*>(&lsB[c][kb2]);
            bfg.u[1] = *reinterpret_cast<const uint4*>(&lsB[c][kb2 + 16]);
            acc[t] = __builtin_amdgcn_wmma_f32_16x16x32_bf16(false, af.v, false, bfg.v,
                                                             (short)0, acc[t], false, false);
        }
        __syncthreads();
    }
    const int rbase = m0 + 16 * wave + 8 * (lane >> 4);
    #pragma unroll
    for (int t = 0; t < 4; ++t) {
        int col = n0 + 16 * t + (lane & 15);
        float bv = bias ? bias[col] : 0.f;
        #pragma unroll
        for (int v = 0; v < 8; ++v) {
            float x = acc[t][v] + bv;
            if (ACT == 1) x = 1.f / (1.f + __expf(-x));
            size_t idx = (size_t)(rbase + v) * Nn + col;
            if (OBF) ((bf16_t*)Out)[idx] = (bf16_t)x;
            else     ((float*)Out)[idx]  = x;
        }
    }
}

// ---------------- adaLN combine: a = sigmoid(gamma) * ln(x) + beta -> bf16 ----------------
__global__ void ew_ada_k(const float* __restrict__ g, const float* __restrict__ lnx,
                         const float* __restrict__ b, bf16_t* __restrict__ out, int n) {
    int i = blockIdx.x * 256 + threadIdx.x;
    if (i < n) out[i] = (bf16_t)(g[i] * lnx[i] + b[i]);
}

// ---------------- SwiGLU: hh = silu(h1)*h2 -> bf16 ----------------
__global__ void ew_swiglu_k(const float* __restrict__ h1, const float* __restrict__ h2,
                            bf16_t* __restrict__ out, int n) {
    int i = blockIdx.x * 256 + threadIdx.x;
    if (i < n) {
        float a = h1[i];
        float s = a / (1.f + __expf(-a));
        out[i] = (bf16_t)(s * h2[i]);
    }
}

// ---------------- final residual: out = x + gA*ao + gF*ffo ----------------
__global__ void ew_final_k(const float* __restrict__ x, const float* __restrict__ gA,
                           const float* __restrict__ ao, const float* __restrict__ gF,
                           const float* __restrict__ ffo, float* __restrict__ out, int n) {
    int i = blockIdx.x * 256 + threadIdx.x;
    if (i < n) out[i] = x[i] + gA[i] * ao[i] + gF[i] * ffo[i];
}

// ---------------- pair bias (WMMA version): one streaming pass over pair_cond ----------------
// bias[h][i][j] = (sum_p x_p*wb[p][h] - mean * colsum_h) * rsqrt(var+eps)
__global__ __launch_bounds__(128)
void pair_bias_k(const float* __restrict__ pair, const float* __restrict__ wb,
                 float* __restrict__ bias) {
    __shared__ alignas(16) bf16_t lsX[128][136];   // rows padded: 136*2B = 272B (16B mult)
    __shared__ alignas(16) bf16_t lsW[16][136];    // wb transposed: [h][p]
    __shared__ float lsM[128][2];                  // mean, inv per row
    __shared__ float scol[16];
    const int tid = threadIdx.x;

    for (int i = tid; i < DP_ * H_; i += 128) lsW[i & 15][i >> 4] = (bf16_t)wb[i];
    if (tid < H_) {
        float s = 0.f;
        for (int p = 0; p < DP_; ++p) s += wb[p * H_ + tid];
        scol[tid] = s;
    }

    const size_t pidx = (size_t)blockIdx.x * 128 + tid;
    const float* xp = pair + pidx * DP_;
    __builtin_prefetch(xp, 0, 3);
    __builtin_prefetch(xp + 32, 0, 3);
    __builtin_prefetch(xp + 64, 0, 3);
    __builtin_prefetch(xp + 96, 0, 3);
    float S = 0.f, Q = 0.f;
    for (int c = 0; c < DP_; c += 8) {
        float4 v0 = *reinterpret_cast<const float4*>(xp + c);
        float4 v1 = *reinterpret_cast<const float4*>(xp + c + 4);
        union { bf16_t b[8]; uint4 u; } pk;
        float e[8] = {v0.x, v0.y, v0.z, v0.w, v1.x, v1.y, v1.z, v1.w};
        #pragma unroll
        for (int u = 0; u < 8; ++u) { S += e[u]; Q += e[u] * e[u]; pk.b[u] = (bf16_t)e[u]; }
        *reinterpret_cast<uint4*>(&lsX[tid][c]) = pk.u;
    }
    {
        float mean = S * (1.f / DP_);
        float var  = Q * (1.f / DP_) - mean * mean;
        lsM[tid][0] = mean;
        lsM[tid][1] = rsqrtf(var + 1e-5f);
    }
    __syncthreads();

    const int wave = tid >> 5, lane = tid & 31;
    AFrag bw[4];
    #pragma unroll
    for (int ks = 0; ks < 4; ++ks) {
        int hcol = lane & 15, kb2 = 32 * ks + 8 * (lane >> 4);
        bw[ks].u[0] = *reinterpret_cast<const uint4*>(&lsW[hcol][kb2]);
        bw[ks].u[1] = *reinterpret_cast<const uint4*>(&lsW[hcol][kb2 + 16]);
    }
    #pragma unroll
    for (int tt = 0; tt < 2; ++tt) {
        const int tile = wave * 2 + tt;       // 0..7 -> 16 rows each
        const int rbase = tile * 16;
        v8f acc;
        for (int v = 0; v < 8; ++v) acc[v] = 0.f;
        #pragma unroll
        for (int ks = 0; ks < 4; ++ks) {
            AFrag ax;
            int r = rbase + (lane & 15), kh = lane >> 4;
            ax.u[0] = *reinterpret_cast<const uint4*>(&lsX[r][32 * ks + 8 * kh]);
            ax.u[1] = *reinterpret_cast<const uint4*>(&lsX[r][32 * ks + 8 * kh + 16]);
            acc = __builtin_amdgcn_wmma_f32_16x16x32_bf16(false, ax.v, false, bw[ks].v,
                                                          (short)0, acc, false, false);
        }
        const int head = lane & 15;
        const float sc_h = scol[head];
        #pragma unroll
        for (int v = 0; v < 8; ++v) {
            int lr = rbase + 8 * (lane >> 4) + v;
            float mean = lsM[lr][0], inv = lsM[lr][1];
            size_t pp = (size_t)blockIdx.x * 128 + lr;
            int i = (int)(pp >> 10), j = (int)(pp & 1023);
            bias[((size_t)head << 20) + ((size_t)i << 10) + j] = (acc[v] - mean * sc_h) * inv;
        }
    }
}

// ---------------- fused flash attention per (head, 16-row tile), one wave/block ----------------
__global__ __launch_bounds__(32)
void attn_k(const bf16_t* __restrict__ qb, const bf16_t* __restrict__ kb,
            const bf16_t* __restrict__ vb, const float* __restrict__ bias,
            const float* __restrict__ gsig, const unsigned char* __restrict__ mask,
            bf16_t* __restrict__ go) {
    const int h    = blockIdx.y;
    const int i0   = blockIdx.x * 16;
    const int lane = threadIdx.x;
    __shared__ alignas(16) bf16_t lsQ[16][64];
    __shared__ alignas(16) bf16_t lsK[32][64];
    __shared__ alignas(16) bf16_t lsV[48][32];   // transposed: [dh][j]
    __shared__ alignas(16) bf16_t lsP[16][32];

    if (lane < 16) {
        const bf16_t* src = qb + (size_t)(i0 + lane) * DS_ + h * DH_;
        #pragma unroll
        for (int c = 0; c < 48; c += 8) copy16B_g2l(src + c, &lsQ[lane][c]);
        #pragma unroll
        for (int c = 48; c < 64; ++c) lsQ[lane][c] = (bf16_t)0.f;
    }
    async_wait_all();
    __syncthreads();
    AFrag aq0, aq1;
    {
        int r = lane & 15, kh = lane >> 4;
        aq0.u[0] = *reinterpret_cast<const uint4*>(&lsQ[r][8 * kh]);
        aq0.u[1] = *reinterpret_cast<const uint4*>(&lsQ[r][8 * kh + 16]);
        aq1.u[0] = *reinterpret_cast<const uint4*>(&lsQ[r][32 + 8 * kh]);
        aq1.u[1] = *reinterpret_cast<const uint4*>(&lsQ[r][32 + 8 * kh + 16]);
    }
    float mrow[8], lrow[8];
    v8f o[3];
    #pragma unroll
    for (int v = 0; v < 8; ++v) { mrow[v] = -1e30f; lrow[v] = 0.f; }
    for (int t = 0; t < 3; ++t) for (int v = 0; v < 8; ++v) o[t][v] = 0.f;
    const float scale = 0.14433756729740643f;  // 1/sqrt(48)

    for (int j0 = 0; j0 < TOK; j0 += 32) {
        {   // K chunk row-major (zero-pad dh 48..63), V chunk transposed
            const bf16_t* ksrc = kb + (size_t)(j0 + lane) * DS_ + h * DH_;
            #pragma unroll
            for (int c = 0; c < 48; c += 8) copy16B_g2l(ksrc + c, &lsK[lane][c]);
            #pragma unroll
            for (int c = 48; c < 64; ++c) lsK[lane][c] = (bf16_t)0.f;
            const bf16_t* vsrc = vb + (size_t)(j0 + lane) * DS_ + h * DH_;
            #pragma unroll
            for (int d = 0; d < 48; ++d) lsV[d][lane] = vsrc[d];
        }
        async_wait_all();
        __syncthreads();
        v8f s[2];
        #pragma unroll
        for (int jt = 0; jt < 2; ++jt) {
            for (int v = 0; v < 8; ++v) s[jt][v] = 0.f;
            #pragma unroll
            for (int ks = 0; ks < 2; ++ks) {
                AFrag bk;
                int c = 16 * jt + (lane & 15), kbase = 32 * ks + 8 * (lane >> 4);
                bk.u[0] = *reinterpret_cast<const uint4*>(&lsK[c][kbase]);
                bk.u[1] = *reinterpret_cast<const uint4*>(&lsK[c][kbase + 16]);
                const AFrag& aq = ks ? aq1 : aq0;
                s[jt] = __builtin_amdgcn_wmma_f32_16x16x32_bf16(false, aq.v, false, bk.v,
                                                                (short)0, s[jt], false, false);
            }
            int j = j0 + 16 * jt + (lane & 15);
            float mko = mask[j] ? 0.f : -1e9f;
            const float* bp = bias + ((size_t)h << 20) + ((size_t)(i0 + 8 * (lane >> 4)) << 10) + j;
            #pragma unroll
            for (int v = 0; v < 8; ++v)
                s[jt][v] = s[jt][v] * scale + bp[(size_t)v << 10] + mko;
        }
        // online softmax (reductions stay within 16-lane halves: xor masks 1,2,4,8)
        #pragma unroll
        for (int v = 0; v < 8; ++v) {
            float t = fmaxf(s[0][v], s[1][v]);
            for (int m = 1; m < 16; m <<= 1) t = fmaxf(t, __shfl_xor(t, m, 32));
            float mn = fmaxf(mrow[v], t);
            float al = __expf(mrow[v] - mn);
            mrow[v] = mn;
            float p0 = __expf(s[0][v] - mn), p1 = __expf(s[1][v] - mn);
            float rs = p0 + p1;
            for (int m = 1; m < 16; m <<= 1) rs += __shfl_xor(rs, m, 32);
            lrow[v] = lrow[v] * al + rs;
            #pragma unroll
            for (int t2 = 0; t2 < 3; ++t2) o[t2][v] *= al;
            int row = 8 * (lane >> 4) + v, col = lane & 15;
            lsP[row][col]      = (bf16_t)p0;
            lsP[row][16 + col] = (bf16_t)p1;
        }
        __syncthreads();
        AFrag ap;
        {
            int r = lane & 15, kh = lane >> 4;
            ap.u[0] = *reinterpret_cast<const uint4*>(&lsP[r][8 * kh]);
            ap.u[1] = *reinterpret_cast<const uint4*>(&lsP[r][8 * kh + 16]);
        }
        #pragma unroll
        for (int t = 0; t < 3; ++t) {
            AFrag bv;
            int d = 16 * t + (lane & 15), kbase = 8 * (lane >> 4);
            bv.u[0] = *reinterpret_cast<const uint4*>(&lsV[d][kbase]);
            bv.u[1] = *reinterpret_cast<const uint4*>(&lsV[d][kbase + 16]);
            o[t] = __builtin_amdgcn_wmma_f32_16x16x32_bf16(false, ap.v, false, bv.v,
                                                           (short)0, o[t], false, false);
        }
        __syncthreads();
    }
    #pragma unroll
    for (int t = 0; t < 3; ++t) {
        int col = h * DH_ + 16 * t + (lane & 15);
        #pragma unroll
        for (int v = 0; v < 8; ++v) {
            int row = i0 + 8 * (lane >> 4) + v;
            float val = o[t][v] / lrow[v];
            float gv  = gsig[(size_t)row * DS_ + col];
            go[(size_t)row * DS_ + col] = (bf16_t)(val * gv);
        }
    }
}

// ---------------- host side ----------------
static inline size_t align_up256(size_t x) { return (x + 255) & ~(size_t)255; }

extern "C" void kernel_launch(void* const* d_in, const int* in_sizes, int n_in,
                              void* d_out, int out_size, void* d_ws, size_t ws_size,
                              hipStream_t stream) {
    (void)in_sizes; (void)n_in; (void)out_size; (void)ws_size;
    const float* x    = (const float*)d_in[0];
    const float* sc   = (const float*)d_in[1];
    const float* pair = (const float*)d_in[2];
    const float* agw  = (const float*)d_in[3];
    const float* agb  = (const float*)d_in[4];
    const float* abw  = (const float*)d_in[5];
    const float* wq   = (const float*)d_in[6];
    const float* bq   = (const float*)d_in[7];
    const float* wk   = (const float*)d_in[8];
    const float* wv   = (const float*)d_in[9];
    const float* wb   = (const float*)d_in[10];
    const float* wg   = (const float*)d_in[11];
    const float* wo   = (const float*)d_in[12];
    const float* gtw  = (const float*)d_in[13];
    const float* gtb  = (const float*)d_in[14];
    const float* fgw  = (const float*)d_in[15];
    const float* fgb  = (const float*)d_in[16];
    const float* fbw  = (const float*)d_in[17];
    const float* fw1  = (const float*)d_in[18];
    const float* fw2  = (const float*)d_in[19];
    const float* fw3  = (const float*)d_in[20];
    const float* fgtw = (const float*)d_in[21];
    const float* fgtb = (const float*)d_in[22];
    const unsigned char* mask = (const unsigned char*)d_in[23];

    const int DS2 = DS_ * DS_;         // 589824
    const int DSF = DS_ * DFF_;        // 1179648
    const int ND  = TOK * DS_;         // 786432
    const int NF  = TOK * DFF_;        // 1572864

    char* w = (char*)d_ws;
    size_t off = 0;
    auto alloc = [&](size_t bytes) -> void* {
        void* p = w + off;
        off = align_up256(off + bytes);
        return p;
    };

    // bf16 weights
    bf16_t* b_agw  = (bf16_t*)alloc((size_t)DS2 * 2);
    bf16_t* b_abw  = (bf16_t*)alloc((size_t)DS2 * 2);
    bf16_t* b_wq   = (bf16_t*)alloc((size_t)DS2 * 2);
    bf16_t* b_wk   = (bf16_t*)alloc((size_t)DS2 * 2);
    bf16_t* b_wv   = (bf16_t*)alloc((size_t)DS2 * 2);
    bf16_t* b_wg   = (bf16_t*)alloc((size_t)DS2 * 2);
    bf16_t* b_wo   = (bf16_t*)alloc((size_t)DS2 * 2);
    bf16_t* b_gtw  = (bf16_t*)alloc((size_t)DS2 * 2);
    bf16_t* b_fgw  = (bf16_t*)alloc((size_t)DS2 * 2);
    bf16_t* b_fbw  = (bf16_t*)alloc((size_t)DS2 * 2);
    bf16_t* b_fgtw = (bf16_t*)alloc((size_t)DS2 * 2);
    bf16_t* b_w1   = (bf16_t*)alloc((size_t)DSF * 2);
    bf16_t* b_w2   = (bf16_t*)alloc((size_t)DSF * 2);
    bf16_t* b_w3   = (bf16_t*)alloc((size_t)DSF * 2);
    // activations
    bf16_t* sn_bf  = (bf16_t*)alloc((size_t)ND * 2);
    bf16_t* sc_bf  = (bf16_t*)alloc((size_t)ND * 2);
    float*  lnx    = (float*)alloc((size_t)ND * 4);
    float*  tmpg   = (float*)alloc((size_t)ND * 4);
    float*  tmpb   = (float*)alloc((size_t)ND * 4);
    bf16_t* a_bf   = (bf16_t*)alloc((size_t)ND * 2);
    bf16_t* qb     = (bf16_t*)alloc((size_t)ND * 2);
    bf16_t* kb     = (bf16_t*)alloc((size_t)ND * 2);
    bf16_t* vbuf   = (bf16_t*)alloc((size_t)ND * 2);
    float*  gsig   = (float*)alloc((size_t)ND * 4);
    float*  gateA  = (float*)alloc((size_t)ND * 4);
    float*  gateF  = (float*)alloc((size_t)ND * 4);
    bf16_t* f_bf   = (bf16_t*)alloc((size_t)ND * 2);
    float*  h1     = (float*)alloc((size_t)NF * 4);
    float*  h2     = (float*)alloc((size_t)NF * 4);
    bf16_t* hh_bf  = (bf16_t*)alloc((size_t)NF * 2);
    float*  ffo    = (float*)alloc((size_t)ND * 4);
    bf16_t* go     = (bf16_t*)alloc((size_t)ND * 2);
    float*  aout   = (float*)alloc((size_t)ND * 4);
    float*  biasT  = (float*)alloc((size_t)H_ * TOK * TOK * 4);   // 64 MB

    auto cvt = [&](const float* src, bf16_t* dst, int n) {
        cvt_f32_bf16_k<<<(n + 255) / 256, 256, 0, stream>>>(src, dst, n);
    };
    auto gemm = [&](const bf16_t* A, const bf16_t* B, const float* bias, void* out,
                    int M, int Nn, int K, int act, int obf) {
        dim3 g(Nn / 64, M / 128);
        if (act == 1)      gemm_bf16_k<1, 0><<<g, 256, 0, stream>>>(A, B, bias, out, M, Nn, K);
        else if (obf == 1) gemm_bf16_k<0, 1><<<g, 256, 0, stream>>>(A, B, bias, out, M, Nn, K);
        else               gemm_bf16_k<0, 0><<<g, 256, 0, stream>>>(A, B, bias, out, M, Nn, K);
    };

    // weight conversion
    cvt(agw, b_agw, DS2);  cvt(abw, b_abw, DS2);
    cvt(wq, b_wq, DS2);    cvt(wk, b_wk, DS2);   cvt(wv, b_wv, DS2);
    cvt(wg, b_wg, DS2);    cvt(wo, b_wo, DS2);   cvt(gtw, b_gtw, DS2);
    cvt(fgw, b_fgw, DS2);  cvt(fbw, b_fbw, DS2); cvt(fgtw, b_fgtw, DS2);
    cvt(fw1, b_w1, DSF);   cvt(fw2, b_w2, DSF);  cvt(fw3, b_w3, DSF);
    // LN(single_cond) -> bf16 ; LN(x) -> f32 ; raw single_cond -> bf16
    ln_rows_k<<<TOK, 256, 0, stream>>>(sc, sn_bf, nullptr, DS_);
    ln_rows_k<<<TOK, 256, 0, stream>>>(x, nullptr, lnx, DS_);
    cvt(sc, sc_bf, ND);

    // ---- attention branch ----
    gemm(sn_bf, b_agw, agb, tmpg, TOK, DS_, DS_, /*sig*/1, 0);
    gemm(sn_bf, b_abw, nullptr, tmpb, TOK, DS_, DS_, 0, 0);
    ew_ada_k<<<(ND + 255) / 256, 256, 0, stream>>>(tmpg, lnx, tmpb, a_bf, ND);
    gemm(a_bf, b_wq, bq, qb, TOK, DS_, DS_, 0, /*bf16*/1);
    gemm(a_bf, b_wk, nullptr, kb, TOK, DS_, DS_, 0, 1);
    gemm(a_bf, b_wv, nullptr, vbuf, TOK, DS_, DS_, 0, 1);
    gemm(a_bf, b_wg, nullptr, gsig, TOK, DS_, DS_, 1, 0);
    gemm(sc_bf, b_gtw, gtb, gateA, TOK, DS_, DS_, 1, 0);
    pair_bias_k<<<(TOK * TOK) / 128, 128, 0, stream>>>(pair, wb, biasT);
    attn_k<<<dim3(TOK / 16, H_), 32, 0, stream>>>(qb, kb, vbuf, biasT, gsig, mask, go);
    gemm(go, b_wo, nullptr, aout, TOK, DS_, DS_, 0, 0);

    // ---- feed-forward branch ----
    gemm(sn_bf, b_fgw, fgb, tmpg, TOK, DS_, DS_, 1, 0);
    gemm(sn_bf, b_fbw, nullptr, tmpb, TOK, DS_, DS_, 0, 0);
    ew_ada_k<<<(ND + 255) / 256, 256, 0, stream>>>(tmpg, lnx, tmpb, f_bf, ND);
    gemm(f_bf, b_w1, nullptr, h1, TOK, DFF_, DS_, 0, 0);
    gemm(f_bf, b_w2, nullptr, h2, TOK, DFF_, DS_, 0, 0);
    ew_swiglu_k<<<(NF + 255) / 256, 256, 0, stream>>>(h1, h2, hh_bf, NF);
    gemm(hh_bf, b_w3, nullptr, ffo, TOK, DS_, DFF_, 0, 0);
    gemm(sc_bf, b_fgtw, fgtb, gateF, TOK, DS_, DS_, 1, 0);

    // ---- residual combine ----
    ew_final_k<<<(ND + 255) / 256, 256, 0, stream>>>(x, gateA, aout, gateF, ffo,
                                                     (float*)d_out, ND);
}
